// RNN_89515708383967
// MI455X (gfx1250) — matrix operand order repaired and checked by reading, not compiled, not run on hardware
//
#include <hip/hip_runtime.h>
#include <math.h>

typedef __attribute__((ext_vector_type(16))) __bf16 v16bf;
typedef __attribute__((ext_vector_type(8)))  __bf16 v8bf;
typedef __attribute__((ext_vector_type(8)))  float  v8f;

constexpr int T = 512, B = 64, I = 512, H = 512;
constexpr int TB = T * B;            // 32768 rows in the hoisted GEMM
constexpr int MT = TB / 16;          // 2048 M tiles
constexpr int NT = H / 16;           // 32 N tiles
constexpr int NWG_SEQ = 32;          // workgroups in the persistent recurrent kernel

// ---------------------------------------------------------------------------
// Kernel 1: convert weights f32 -> bf16 into workspace; reset grid barrier.
// ---------------------------------------------------------------------------
__global__ void cvt_weights(const float* __restrict__ wih,
                            const float* __restrict__ whh,
                            __bf16* __restrict__ wih_b,
                            __bf16* __restrict__ whh_b,
                            unsigned* __restrict__ counter) {
    int idx    = blockIdx.x * blockDim.x + threadIdx.x;
    int stride = gridDim.x * blockDim.x;
    const int n1 = H * I, ntot = H * I + H * H;
    for (int i = idx; i < ntot; i += stride) {
        if (i < n1) wih_b[i]      = (__bf16)wih[i];
        else        whh_b[i - n1] = (__bf16)whh[i - n1];
    }
    if (idx == 0) *counter = 0u;     // re-armed every launch -> replay-safe
}

// ---------------------------------------------------------------------------
// Kernel 2: pre = x @ w_ih^T  (32768 x 512 x 512), output in C-FRAGMENT order:
//   pre[(mt*NT + nt)*256 + lane*8 + r]  == C VGPR r of lane `lane` of tile
// Each wave computes 4 N-tiles sharing one A fragment (4x less x traffic).
// 8 waves/WG provide wave-level latency hiding here.
// ---------------------------------------------------------------------------
__global__ void __launch_bounds__(256) gemm_ih(const float* __restrict__ x,
                                               const __bf16* __restrict__ wb,
                                               float* __restrict__ pre) {
    const int lane = threadIdx.x & 31;
    const int wid  = threadIdx.x >> 5;
    const int g    = blockIdx.x * 8 + wid;   // global wave id, 0..16383
    const int mt   = g >> 3;                 // 0..2047 (M tile)
    const int nb   = g & 7;                  // N-tile block: tiles nb*4 .. nb*4+3
    const int ml   = lane & 15;
    const int half = lane >> 4;

    const float* arow = x + (size_t)(mt * 16 + ml) * I;
    const __bf16* brow0 = wb + (size_t)((nb * 4 + 0) * 16 + ml) * I;
    const __bf16* brow1 = wb + (size_t)((nb * 4 + 1) * 16 + ml) * I;
    const __bf16* brow2 = wb + (size_t)((nb * 4 + 2) * 16 + ml) * I;
    const __bf16* brow3 = wb + (size_t)((nb * 4 + 3) * 16 + ml) * I;

    v8f acc0 = {}, acc1 = {}, acc2 = {}, acc3 = {};
#pragma unroll 4
    for (int k0 = 0; k0 < I; k0 += 32) {
        // A fragment: K = k0 + 16*half + [0..15]
        v8f af0 = *(const v8f*)(arow + k0 + 16 * half);
        v8f af1 = *(const v8f*)(arow + k0 + 16 * half + 8);
        v16bf a;
#pragma unroll
        for (int i = 0; i < 8; ++i) { a[i] = (__bf16)af0[i]; a[i + 8] = (__bf16)af1[i]; }
        const int kb = k0 + 16 * half;       // B fragment: K = kb + [0..15]
        v16bf b0 = *(const v16bf*)(brow0 + kb);
        v16bf b1 = *(const v16bf*)(brow1 + kb);
        v16bf b2 = *(const v16bf*)(brow2 + kb);
        v16bf b3 = *(const v16bf*)(brow3 + kb);
        acc0 = __builtin_amdgcn_wmma_f32_16x16x32_bf16(false, a, false, b0, (short)0, acc0, false, false);
        acc1 = __builtin_amdgcn_wmma_f32_16x16x32_bf16(false, a, false, b1, (short)0, acc1, false, false);
        acc2 = __builtin_amdgcn_wmma_f32_16x16x32_bf16(false, a, false, b2, (short)0, acc2, false, false);
        acc3 = __builtin_amdgcn_wmma_f32_16x16x32_bf16(false, a, false, b3, (short)0, acc3, false, false);
    }
    // Store in fragment order: one contiguous 32B store per lane per tile.
    float* p = pre + ((size_t)mt * NT + nb * 4) * 256 + lane * 8;
    *(v8f*)(p)       = acc0;
    *(v8f*)(p + 256) = acc1;
    *(v8f*)(p + 512) = acc2;
    *(v8f*)(p + 768) = acc3;
}

// ---------------------------------------------------------------------------
// Kernel 3: persistent sequential recurrence.
// 32 WGs x 4 waves = 128 waves = the 4x32 tiles of the 64x512 h_t.
// w_hh B-fragments live in LDS (staged once). The K-loop is software-
// pipelined: chunk k+1's A loads are issued before chunk k's WMMA, so the
// L2 load latency overlaps ds_load+WMMA instead of stalling the chain
// (only 1 wave/SIMD here -> no wave-level hiding). pre[t+1] is prefetched
// across the barrier since it doesn't depend on h_t.
// ---------------------------------------------------------------------------
__global__ void __launch_bounds__(128) rnn_seq(const float* __restrict__ pre,
                                               const __bf16* __restrict__ whh,
                                               __bf16* __restrict__ hb0,
                                               __bf16* __restrict__ hb1,
                                               float* __restrict__ out,
                                               unsigned* __restrict__ counter) {
    __shared__ __bf16 lwb[4 * 16 * H];       // 64 KB: 4 waves x 16 w_hh rows

    const int lane = threadIdx.x & 31;
    const int wid  = threadIdx.x >> 5;       // 0..3
    const int g    = blockIdx.x * 4 + wid;   // 0..127
    const int mt   = g >> 5;                 // 0..3  (batch tile)
    const int nt   = g & 31;                 // 0..31 (hidden tile)
    const int ml   = lane & 15;
    const int half = lane >> 4;
    const int n     = nt * 16 + ml;          // output column for B/C fragments
    const int arow  = mt * 16 + ml;          // batch row for A fragment
    const int crow0 = mt * 16 + 8 * half;    // first C/D row for this lane half

    // ---- stage w_hh rows [nt*16, nt*16+16) into LDS (16KB contiguous) ----
    {
        const uint4* src = (const uint4*)(whh + (size_t)nt * 16 * H);
        uint4* dst = (uint4*)(lwb + (size_t)wid * 16 * H);
#pragma unroll 8
        for (int i = lane; i < 16 * H * (int)sizeof(__bf16) / 16; i += 32)
            dst[i] = src[i];
    }
    __syncthreads();
    const __bf16* lbrow = lwb + (size_t)(wid * 16 + ml) * H;  // B col n, in LDS

    // Prefetch the first step's pre fragment (fragment order -> one 32B load).
    v8f pnext = *(const v8f*)(pre + ((size_t)(0 * 4 + mt) * NT + nt) * 256 + lane * 8);

    for (int t = 0; t < T; ++t) {
        const __bf16* hprev = (t & 1) ? hb1 : hb0;   // h_{t-1} lives in buf[t&1]
        __bf16*       hnext = (t & 1) ? hb0 : hb1;   // h_t goes to buf[(t+1)&1]

        v8f acc = pnext;
        // Issue next step's pre load now: it has the whole step to complete.
        if (t + 1 < T)
            pnext = *(const v8f*)(pre + ((size_t)((t + 1) * 4 + mt) * NT + nt) * 256 + lane * 8);

        if (t > 0) {   // t==0: h_prev == 0, skip the recurrent GEMM entirely
            const __bf16* ar = hprev + (size_t)arow * H;
            // prologue: A fragment for chunk 0
            v8bf a0 = *(const v8bf*)(ar + 16 * half);
            v8bf a1 = *(const v8bf*)(ar + 8 + 16 * half);
#pragma unroll
            for (int kc = 0; kc < 16; ++kc) {
                const int k0 = kc * 32;
                v8bf na0 = a0, na1 = a1;
                if (kc < 15) {   // prefetch chunk kc+1 before this chunk's WMMA
                    na0 = *(const v8bf*)(ar + k0 + 32 + 16 * half);
                    na1 = *(const v8bf*)(ar + k0 + 40 + 16 * half);
                }
                v16bf a = __builtin_shufflevector(a0, a1,
                          0, 1, 2, 3, 4, 5, 6, 7, 8, 9, 10, 11, 12, 13, 14, 15);
                v16bf b = *(const v16bf*)(lbrow + k0 + 16 * half);   // ds_load
                acc = __builtin_amdgcn_wmma_f32_16x16x32_bf16(false, a, false, b,
                                                              (short)0, acc, false, false);
                a0 = na0; a1 = na1;
            }
        }

        // Nonlinearity + stores: f32 h_all, bf16 h for the next step, f32 h_last
        float* ot = out + (size_t)t * (B * H);
#pragma unroll
        for (int r = 0; r < 8; ++r) {
            float v = tanhf(acc[r]);
            size_t idx = (size_t)(crow0 + r) * H + n;
            ot[idx]    = v;
            hnext[idx] = (__bf16)v;
            if (t == T - 1) out[(size_t)T * B * H + idx] = v;
        }

        // ---- device-wide barrier (release stores, signal, spin, acquire) ----
        __threadfence();
        __syncthreads();
        if (threadIdx.x == 0) {
            atomicAdd(counter, 1u);
            const unsigned target = (unsigned)(NWG_SEQ * (t + 1));
            while (atomicAdd(counter, 0u) < target) { __builtin_amdgcn_s_sleep(2); }
        }
        __syncthreads();
        __threadfence();
    }
}

// ---------------------------------------------------------------------------
// Launch: ws = [ pre f32 64MB (fragment order) | w_ih bf16 512KB |
//                w_hh bf16 512KB | h_buf0 64KB | h_buf1 64KB | counter ]
// ---------------------------------------------------------------------------
extern "C" void kernel_launch(void* const* d_in, const int* in_sizes, int n_in,
                              void* d_out, int out_size, void* d_ws, size_t ws_size,
                              hipStream_t stream) {
    (void)in_sizes; (void)n_in; (void)out_size; (void)ws_size;
    const float* x    = (const float*)d_in[0];   // (T, B, I) f32
    const float* w_ih = (const float*)d_in[1];   // (H, I)    f32
    const float* w_hh = (const float*)d_in[2];   // (H, H)    f32
    float* out = (float*)d_out;                  // (T,B,H) ++ (B,H) f32

    char* ws = (char*)d_ws;
    constexpr size_t PRE_BYTES = (size_t)T * B * H * sizeof(float);   // 64 MB
    float*    pre   = (float*)ws;
    __bf16*   wih_b = (__bf16*)(ws + PRE_BYTES);
    __bf16*   whh_b = wih_b + (size_t)H * I;
    __bf16*   hb0   = whh_b + (size_t)H * H;
    __bf16*   hb1   = hb0 + (size_t)B * H;
    unsigned* cnt   = (unsigned*)(hb1 + (size_t)B * H);

    // 1) bf16 weight staging + barrier reset (replay-deterministic)
    cvt_weights<<<512, 256, 0, stream>>>(w_ih, w_hh, wih_b, whh_b, cnt);

    // 2) hoisted input projection -> pre (fragment layout)
    gemm_ih<<<MT, 256, 0, stream>>>(x, wih_b, pre);

    // 3) persistent sequential recurrence over T steps
    rnn_seq<<<NWG_SEQ, 128, 0, stream>>>(pre, whh_b, hb0, hb1, out, cnt);
}
